// LittleBitOnDeviceLinear_63342177681876
// MI455X (gfx1250) — compile-verified, matrix operand order checked by reading, not run on hardware
//
#include <hip/hip_runtime.h>
#include <hip/hip_bf16.h>

typedef __attribute__((ext_vector_type(16))) __bf16 bf16x16;
typedef __attribute__((ext_vector_type(8)))  __bf16 bf16x8;
typedef __attribute__((ext_vector_type(4)))  __bf16 bf16x4;
typedef __attribute__((ext_vector_type(8)))  float  f32x8;

// ---------------- Tiling config ----------------
// Block tile 128x128, K-tile 64 (two WMMA K-steps -> 16 WMMAs per barrier pair).
// 256 threads = 8 waves (wave32). Wave grid 4x2: each wave owns 32 rows x 64 cols
// = 2x4 fragments of 16x16. Next K-tile is prefetched into VGPRs during compute.
#define BM 128
#define BN 128
#define BK 64
#define LDB 72   // 64 + 8 bf16 pad -> 144B row stride (16B aligned)

// -------- quantize x -> bf16 (v2 is folded into the sign weights) --------
__global__ void k_quant_bf(const float* __restrict__ x, __bf16* __restrict__ out, long n4)
{
    long i = (long)blockIdx.x * blockDim.x + threadIdx.x;
    if (i >= n4) return;
    float4 xv = ((const float4*)x)[i];
    bf16x4 o;
    o[0] = (__bf16)xv.x; o[1] = (__bf16)xv.y;
    o[2] = (__bf16)xv.z; o[3] = (__bf16)xv.w;
    *(bf16x4*)(out + i * 4) = o;
}

// -------- sign(W)*colscale[k] -> bf16 (folds v2 into Vq) --------
__global__ void k_sign_scale(const float* __restrict__ w, const float* __restrict__ cs,
                             __bf16* __restrict__ out, int K, long n4)
{
    long i = (long)blockIdx.x * blockDim.x + threadIdx.x;
    if (i >= n4) return;
    long e = i * 4;
    int  k = (int)(e % (long)K);
    float4 v = ((const float4*)w)[i];
    float4 s = ((const float4*)cs)[k >> 2];
    bf16x4 o;
    o[0] = (__bf16)((v.x > 0.f) ? s.x : ((v.x < 0.f) ? -s.x : 0.f));
    o[1] = (__bf16)((v.y > 0.f) ? s.y : ((v.y < 0.f) ? -s.y : 0.f));
    o[2] = (__bf16)((v.z > 0.f) ? s.z : ((v.z < 0.f) ? -s.z : 0.f));
    o[3] = (__bf16)((v.w > 0.f) ? s.w : ((v.w < 0.f) ? -s.w : 0.f));
    *(bf16x4*)(out + e) = o;
}

// -------- sign(W) -> bf16 (+1/-1/0) --------
__global__ void k_sign(const float* __restrict__ w, __bf16* __restrict__ out, long n4)
{
    long i = (long)blockIdx.x * blockDim.x + threadIdx.x;
    if (i >= n4) return;
    float4 v = ((const float4*)w)[i];
    bf16x4 o;
    o[0] = (__bf16)((v.x > 0.f) ? 1.f : ((v.x < 0.f) ? -1.f : 0.f));
    o[1] = (__bf16)((v.y > 0.f) ? 1.f : ((v.y < 0.f) ? -1.f : 0.f));
    o[2] = (__bf16)((v.z > 0.f) ? 1.f : ((v.z < 0.f) ? -1.f : 0.f));
    o[3] = (__bf16)((v.w > 0.f) ? 1.f : ((v.w < 0.f) ? -1.f : 0.f));
    *(bf16x4*)(out + i * 4) = o;
}

// CDNA5 16-bit A/B 16x32 fragment from LDS:
// lane L: row = row0 + (L&15), half = L>>4;
// elems 0..7  -> K = 8*half + i ; elems 8..15 -> K = 16 + 8*half + (i-8)
__device__ __forceinline__ bf16x16 load_frag(const __bf16* base, int row0, int lane)
{
    const __bf16* p = base + (size_t)(row0 + (lane & 15)) * LDB;
    int h8 = ((lane >> 4) & 1) << 3;
    bf16x8 lo = *(const bf16x8*)(p + h8);
    bf16x8 hi = *(const bf16x8*)(p + 16 + h8);
    union { bf16x16 v; bf16x8 h[2]; } u;
    u.h[0] = lo; u.h[1] = hi;
    return u.v;
}

// Full K mainloop: acc += A[m0:m0+128, :K] @ B[n0:n0+128, :K]^T
// Register-staged double buffer: global loads for tile kt+1 issued before the
// 16 WMMAs of tile kt, so VMEM latency overlaps compute.
__device__ __forceinline__ void gemm_loop(
    const __bf16* __restrict__ A, const __bf16* __restrict__ B,
    long m0, long n0, int K,
    __bf16* lds_a, __bf16* lds_b,
    int tid, int lane, int wm, int wn,
    f32x8 acc[2][4])
{
    const int lrow  = tid >> 1;      // 0..127
    const int lhalf = tid & 1;       // 32-elem half of BK=64
    const __bf16* gA = A + (m0 + lrow) * (long)K + lhalf * 32;
    const __bf16* gB = B + (n0 + lrow) * (long)K + lhalf * 32;
    __bf16* sA = lds_a + lrow * LDB + lhalf * 32;
    __bf16* sB = lds_b + lrow * LDB + lhalf * 32;

    bf16x8 ra[4], rb[4];
    #pragma unroll
    for (int q = 0; q < 4; ++q) {
        ra[q] = *(const bf16x8*)(gA + q * 8);
        rb[q] = *(const bf16x8*)(gB + q * 8);
    }

    for (int kt = 0; kt < K; kt += BK) {
        __syncthreads();   // previous tile's fragment reads complete
        #pragma unroll
        for (int q = 0; q < 4; ++q) {
            *(bf16x8*)(sA + q * 8) = ra[q];
            *(bf16x8*)(sB + q * 8) = rb[q];
        }
        __syncthreads();

        int kn = kt + BK;
        if (kn < K) {      // prefetch next tile into registers (in flight during WMMAs)
            #pragma unroll
            for (int q = 0; q < 4; ++q) {
                ra[q] = *(const bf16x8*)(gA + kn + q * 8);
                rb[q] = *(const bf16x8*)(gB + kn + q * 8);
            }
        }

        #pragma unroll
        for (int s = 0; s < 2; ++s) {   // two 16x16x32 K-steps within BK=64
            bf16x16 af[2], bfr[4];
            #pragma unroll
            for (int i = 0; i < 2; ++i)
                af[i] = load_frag(lds_a + s * 32, wm * 32 + i * 16, lane);
            #pragma unroll
            for (int j = 0; j < 4; ++j)
                bfr[j] = load_frag(lds_b + s * 32, wn * 64 + j * 16, lane);
            #pragma unroll
            for (int i = 0; i < 2; ++i)
                #pragma unroll
                for (int j = 0; j < 4; ++j)
                    acc[i][j] = __builtin_amdgcn_wmma_f32_16x16x32_bf16(
                        false, af[i], false, bfr[j], (short)0, acc[i][j], false, false);
        }
    }
}

// GEMM1: C_bf16[M x N] = (A @ B^T) * (s1[n]*s2[n])
__global__ __launch_bounds__(256)
void k_gemm_s(const __bf16* __restrict__ A, const __bf16* __restrict__ B,
              __bf16* __restrict__ C, int K, int N,
              const float* __restrict__ s1, const float* __restrict__ s2)
{
    __shared__ __align__(16) __bf16 lds_a[BM * LDB];
    __shared__ __align__(16) __bf16 lds_b[BN * LDB];
    const int tid = threadIdx.x, lane = tid & 31, wave = tid >> 5;
    const int wm = wave & 3, wn = wave >> 2;
    const long m0 = (long)blockIdx.y * BM, n0 = (long)blockIdx.x * BN;

    f32x8 acc[2][4];
    #pragma unroll
    for (int i = 0; i < 2; ++i)
        #pragma unroll
        for (int j = 0; j < 4; ++j)
            #pragma unroll
            for (int e = 0; e < 8; ++e) acc[i][j][e] = 0.f;

    gemm_loop(A, B, m0, n0, K, lds_a, lds_b, tid, lane, wm, wn, acc);

    #pragma unroll
    for (int j = 0; j < 4; ++j) {
        long n = n0 + wn * 64 + j * 16 + (lane & 15);
        float sc = s1[n] * s2[n];
        #pragma unroll
        for (int i = 0; i < 2; ++i) {
            long mb = m0 + wm * 32 + i * 16 + (((lane >> 4) & 1) << 3);
            #pragma unroll
            for (int r = 0; r < 8; ++r)
                C[(mb + r) * (long)N + n] = (__bf16)(acc[i][j][r] * sc);
        }
    }
}

// GEMM2 fused over both branches: out = u1[n]*(H1@U1^T) + u1R[n]*(H2@U2^T) + bias[n]
// Single write pass over the 134MB f32 output (no read-modify-write).
__global__ __launch_bounds__(256)
void k_gemm2(const __bf16* __restrict__ H1, const __bf16* __restrict__ U1,
             const __bf16* __restrict__ H2, const __bf16* __restrict__ U2,
             float* __restrict__ C, int K, int N,
             const float* __restrict__ u1, const float* __restrict__ u1R,
             const float* __restrict__ bias)
{
    __shared__ __align__(16) __bf16 lds_a[BM * LDB];
    __shared__ __align__(16) __bf16 lds_b[BN * LDB];
    const int tid = threadIdx.x, lane = tid & 31, wave = tid >> 5;
    const int wm = wave & 3, wn = wave >> 2;
    const long m0 = (long)blockIdx.y * BM, n0 = (long)blockIdx.x * BN;

    f32x8 acc[2][4], accP[2][4];
    #pragma unroll
    for (int i = 0; i < 2; ++i)
        #pragma unroll
        for (int j = 0; j < 4; ++j)
            #pragma unroll
            for (int e = 0; e < 8; ++e) acc[i][j][e] = 0.f;

    gemm_loop(H1, U1, m0, n0, K, lds_a, lds_b, tid, lane, wm, wn, acc);

    #pragma unroll
    for (int i = 0; i < 2; ++i)
        #pragma unroll
        for (int j = 0; j < 4; ++j) {
            accP[i][j] = acc[i][j];
            #pragma unroll
            for (int e = 0; e < 8; ++e) acc[i][j][e] = 0.f;
        }

    gemm_loop(H2, U2, m0, n0, K, lds_a, lds_b, tid, lane, wm, wn, acc);

    #pragma unroll
    for (int j = 0; j < 4; ++j) {
        long n = n0 + wn * 64 + j * 16 + (lane & 15);
        float sa = u1[n], sb = u1R[n], bz = bias[n];
        #pragma unroll
        for (int i = 0; i < 2; ++i) {
            long mb = m0 + wm * 32 + i * 16 + (((lane >> 4) & 1) << 3);
            #pragma unroll
            for (int r = 0; r < 8; ++r)
                C[(mb + r) * (long)N + n] = sa * accP[i][j][r] + sb * acc[i][j][r] + bz;
        }
    }
}

extern "C" void kernel_launch(void* const* d_in, const int* in_sizes, int n_in,
                              void* d_out, int out_size, void* d_ws, size_t ws_size,
                              hipStream_t stream)
{
    const float* x    = (const float*)d_in[0];
    const float* V    = (const float*)d_in[1];
    const float* U    = (const float*)d_in[2];
    const float* v2   = (const float*)d_in[3];
    const float* v1   = (const float*)d_in[4];
    const float* u2   = (const float*)d_in[5];
    const float* u1   = (const float*)d_in[6];
    const float* V_R  = (const float*)d_in[7];
    const float* U_R  = (const float*)d_in[8];
    const float* v2_R = (const float*)d_in[9];
    const float* v1_R = (const float*)d_in[10];
    const float* u2_R = (const float*)d_in[11];
    const float* u1_R = (const float*)d_in[12];
    const float* bias = (const float*)d_in[13];

    const int IN  = in_sizes[3];               // 4096
    const int S   = in_sizes[4];               // 1024
    const int OUT = in_sizes[6];               // 4096
    const long Nr = (long)in_sizes[0] / IN;    // 8192

    __bf16* Aq  = (__bf16*)d_ws;                 // Nr*IN     (64MB)
    __bf16* Vq1 = Aq  + (size_t)Nr * IN;         // S*IN      (8MB)
    __bf16* Vq2 = Vq1 + (size_t)S * IN;          // S*IN
    __bf16* Uq1 = Vq2 + (size_t)S * IN;          // OUT*S     (8MB)
    __bf16* Uq2 = Uq1 + (size_t)OUT * S;         // OUT*S
    __bf16* Hq1 = Uq2 + (size_t)OUT * S;         // Nr*S      (16MB)
    __bf16* Hq2 = Hq1 + (size_t)Nr * S;          // Nr*S

    // Quantize activations once; fold v2 / v2_R into the sign weights.
    { long n4 = Nr * IN / 4;
      k_quant_bf<<<(n4 + 255) / 256, 256, 0, stream>>>(x, Aq, n4); }
    { long n4 = (long)S * IN / 4;
      k_sign_scale<<<(n4 + 255) / 256, 256, 0, stream>>>(V,   v2,   Vq1, IN, n4);
      k_sign_scale<<<(n4 + 255) / 256, 256, 0, stream>>>(V_R, v2_R, Vq2, IN, n4); }
    { long n4 = (long)OUT * S / 4;
      k_sign<<<(n4 + 255) / 256, 256, 0, stream>>>(U,   Uq1, n4);
      k_sign<<<(n4 + 255) / 256, 256, 0, stream>>>(U_R, Uq2, n4); }

    // GEMM1 per branch: Hq = (Aq @ Vq^T) * (v1*u2), bf16 out
    dim3 g1(S / BN, Nr / BM);
    k_gemm_s<<<g1, 256, 0, stream>>>(Aq, Vq1, Hq1, IN, S, v1,   u2);
    k_gemm_s<<<g1, 256, 0, stream>>>(Aq, Vq2, Hq2, IN, S, v1_R, u2_R);

    // Fused GEMM2: single pass writes out = u1*P1 + u1R*P2 + bias
    dim3 g2(OUT / BN, Nr / BM);
    k_gemm2<<<g2, 256, 0, stream>>>(Hq1, Uq1, Hq2, Uq2, (float*)d_out, S, OUT,
                                    u1, u1_R, bias);
}
